// MambaBlock_5789615915603
// MI455X (gfx1250) — matrix operand order repaired and checked by reading, not compiled, not run on hardware
//
#include <hip/hip_runtime.h>
#include <hip/hip_bf16.h>

// ---- problem constants (from reference) ----
#define D_MODEL  1024
#define D_STATE  16
#define D_CONV   4
#define DT_RANK  64
#define D_INNER  2048
#define BATCH    2
#define SEQ_LEN  2048
#define MTOT     (BATCH * SEQ_LEN)         // 4096 rows for every GEMM
#define XPROJ_N  (DT_RANK + 2 * D_STATE)   // 96

typedef __attribute__((ext_vector_type(16))) __bf16 v16bf;
typedef __attribute__((ext_vector_type(8)))  float  v8f;

union V16 { v16bf v; unsigned short s[16]; uint4 q[2]; };

__device__ __forceinline__ unsigned f2bf(float f) {
  // round-to-nearest-even f32 -> bf16 (bit pattern in low 16)
  unsigned u = __float_as_uint(f);
  u += 0x7fffu + ((u >> 16) & 1u);
  return u >> 16;
}

// ---------------------------------------------------------------------------
// f32 -> packed bf16 converter (one float4 -> uint2 per thread)
// ---------------------------------------------------------------------------
__global__ __launch_bounds__(256) void f32_to_bf16_kernel(
    const float* __restrict__ src, unsigned short* __restrict__ dst, int n4)
{
  int i = blockIdx.x * 256 + threadIdx.x;
  if (i < n4) {
    float4 v = ((const float4*)src)[i];
    uint2 p;
    p.x = f2bf(v.x) | (f2bf(v.y) << 16);
    p.y = f2bf(v.z) | (f2bf(v.w) << 16);
    ((uint2*)dst)[i] = p;
  }
}

// ---------------------------------------------------------------------------
// Tiled GEMM:  C[M,N] = A[M,K] * W[N,K]^T   (A,W packed bf16; f32 accumulate)
// All shapes/strides are template constants so addresses fold to immediates.
// Block: 256 thr (8 waves). Block tile 128x128, BK=32. Wave tile 64x32.
// Staging in 4 scalar uint4 quads; pointers advanced in place by BK.
// Steady loop: lstore -> barrier -> gload(next) -> 8x wmma -> barrier.
// EPI==1: C = softplus(C + bias[col])  (dt_proj epilogue)
// ---------------------------------------------------------------------------
constexpr int BM = 128, BN = 128, BK = 32, LDSK = 40; // 40-half padded stride

template <int EPI, int NN, int KK, int LDA, int LDW, int LDC>
__global__ __launch_bounds__(256) void gemm_bf16_wmma(
    const unsigned short* __restrict__ A, const unsigned short* __restrict__ W,
    float* __restrict__ C, const float* __restrict__ bias)
{
  __shared__ unsigned short As[BM * LDSK];   // 10240 B
  __shared__ unsigned short Ws[BN * LDSK];   // 10240 B

  const int tid  = threadIdx.x;
  const int lane = tid & 31;
  const int wave = tid >> 5;
  const int wm   = (wave >> 2) * 64;  // wave M offset in block tile (0,64)
  const int wn   = (wave & 3) * 32;   // wave N offset (0,32,64,96)
  const int m0   = blockIdx.y * BM;
  const int n0   = blockIdx.x * BN;

  v8f acc[4][2];
  #pragma unroll
  for (int m = 0; m < 4; ++m)
    #pragma unroll
    for (int n = 0; n < 2; ++n)
      #pragma unroll
      for (int e = 0; e < 8; ++e) acc[m][n][e] = 0.0f;

  // ---- per-thread staging geometry: two 16B chunks per matrix ----
  const int r0 = tid >> 2;            // rows 0..63
  const int r1 = r0 + 64;             // rows 64..127
  const int c8 = (tid & 3) << 3;      // half offset 0,8,16,24
  // W rows beyond NN are clamped to row 0: the garbage only reaches output
  // columns that the (col < NN) store guard discards.
  const int wr0 = (NN % BN == 0) ? (n0 + r0) : ((n0 + r0 < NN) ? n0 + r0 : 0);
  const int wr1 = (NN % BN == 0) ? (n0 + r1) : ((n0 + r1 < NN) ? n0 + r1 : 0);
  const unsigned short* pa0 = A + (size_t)(m0 + r0) * LDA + c8;
  const unsigned short* pa1 = A + (size_t)(m0 + r1) * LDA + c8;
  const unsigned short* pw0 = W + (size_t)wr0 * LDW + c8;
  const unsigned short* pw1 = W + (size_t)wr1 * LDW + c8;
  unsigned short* sa0 = &As[r0 * LDSK + c8];
  unsigned short* sa1 = &As[r1 * LDSK + c8];
  unsigned short* sw0 = &Ws[r0 * LDSK + c8];
  unsigned short* sw1 = &Ws[r1 * LDSK + c8];

  uint4 ra0, ra1, rw0, rw1;

#define GLOAD()                                                       \
  do {                                                                \
    ra0 = *(const uint4*)pa0;  pa0 += BK;                             \
    ra1 = *(const uint4*)pa1;  pa1 += BK;                             \
    rw0 = *(const uint4*)pw0;  pw0 += BK;                             \
    rw1 = *(const uint4*)pw1;  pw1 += BK;                             \
  } while (0)

#define LSTORE()                                                      \
  do {                                                                \
    *(uint4*)sa0 = ra0; *(uint4*)sa1 = ra1;                           \
    *(uint4*)sw0 = rw0; *(uint4*)sw1 = rw1;                           \
  } while (0)

#define COMPUTE()                                                     \
  do {                                                                \
    V16 b0, b1;                                                       \
    { /* B 32x16: lanes0-15 K=0..15, lanes16-31 K=16..31 */           \
      int rB = wn + (lane & 15);                                      \
      int kb = (lane >> 4) << 4;                                      \
      b0.q[0] = *(const uint4*)&Ws[rB * LDSK + kb];                   \
      b0.q[1] = *(const uint4*)&Ws[rB * LDSK + kb + 8];               \
      b1.q[0] = *(const uint4*)&Ws[(rB + 16) * LDSK + kb];            \
      b1.q[1] = *(const uint4*)&Ws[(rB + 16) * LDSK + kb + 8];        \
    }                                                                 \
    _Pragma("unroll")                                                 \
    for (int m = 0; m < 4; ++m) {                                     \
      /* A 16x32: lanes0-15 K=0..15, lanes16-31 K=16..31 */           \
      V16 a;                                                          \
      int rA = wm + m * 16 + (lane & 15);                             \
      int ka = (lane >> 4) << 4;                                      \
      a.q[0] = *(const uint4*)&As[rA * LDSK + ka];                    \
      a.q[1] = *(const uint4*)&As[rA * LDSK + ka + 8];                \
      acc[m][0] = __builtin_amdgcn_wmma_f32_16x16x32_bf16(            \
          false, a.v, false, b0.v, (short)0, acc[m][0], false, false);\
      acc[m][1] = __builtin_amdgcn_wmma_f32_16x16x32_bf16(            \
          false, a.v, false, b1.v, (short)0, acc[m][1], false, false);\
    }                                                                 \
  } while (0)

  constexpr int KT = KK / BK;
  GLOAD();
  for (int kt = 0; kt < KT - 1; ++kt) {
    LSTORE();                          // waits pending loads, fills LDS
    __syncthreads();
    GLOAD();                           // next tile; left in flight over WMMAs
    if (kt + 2 < KT) {                 // pointers now at tile kt+2: prefetch
      __builtin_prefetch(pa0, 0, 1);
      __builtin_prefetch(pw0, 0, 1);
    }
    COMPUTE();                         // 8x v_wmma from LDS
    __syncthreads();                   // all waves done reading LDS
  }
  LSTORE();                            // peeled last tile
  __syncthreads();
  COMPUTE();

#undef GLOAD
#undef LSTORE
#undef COMPUTE

  // ---- epilogue: element e of v8f -> row = e + 8*(lane>=16), col = lane&15
  const int hs = (lane >> 4) << 3;
  #pragma unroll
  for (int m = 0; m < 4; ++m) {
    #pragma unroll
    for (int n = 0; n < 2; ++n) {
      int col = n0 + wn + n * 16 + (lane & 15);
      if (NN % BN == 0 || col < NN) {
        #pragma unroll
        for (int e = 0; e < 8; ++e) {
          float v = acc[m][n][e];
          if (EPI == 1) {                 // dt = softplus(. + bias)
            v += bias[col];
            v = (v > 20.f) ? v : log1pf(__expf(v));
          }
          C[(size_t)(m0 + wm + m * 16 + hs + e) * LDC + col] = v;
        }
      }
    }
  }
}

// ---------------------------------------------------------------------------
// Depthwise causal conv (k=4) + bias + SiLU -> xs (f32 for scan, bf16 for GEMM)
// ---------------------------------------------------------------------------
__global__ __launch_bounds__(256) void conv_silu_kernel(
    const float* __restrict__ xz, const float* __restrict__ conv_w,
    const float* __restrict__ conv_b, float* __restrict__ xs_f,
    unsigned short* __restrict__ xs_h)
{
  int idx = blockIdx.x * 256 + threadIdx.x;   // over MTOT * D_INNER
  int d  = idx & (D_INNER - 1);
  int bl = idx >> 11;                          // b*SEQ_LEN + l
  int l  = bl & (SEQ_LEN - 1);
  float s = conv_b[d];
  #pragma unroll
  for (int j = 0; j < D_CONV; ++j) {
    int lj = l - (D_CONV - 1) + j;
    if (lj >= 0)
      s += conv_w[d * D_CONV + j] *
           xz[(size_t)(bl - (D_CONV - 1) + j) * (2 * D_INNER) + d];
  }
  s = s / (1.f + __expf(-s));                  // SiLU
  xs_f[(size_t)bl * D_INNER + d] = s;
  xs_h[(size_t)bl * D_INNER + d] = (unsigned short)f2bf(s);
}

// ---------------------------------------------------------------------------
// Selective scan: one lane per (b,d) channel; h[16] in registers.
// Fuses +xs*D, *SiLU(z); writes bf16 y ready for the out_proj WMMA GEMM.
// ---------------------------------------------------------------------------
__global__ __launch_bounds__(256) void scan_kernel(
    const float* __restrict__ xdbl, const float* __restrict__ dtb,
    const float* __restrict__ xs,  const float* __restrict__ xz,
    const float* __restrict__ A_log, const float* __restrict__ Dv,
    unsigned short* __restrict__ y)
{
  const int d = blockIdx.x * 256 + threadIdx.x;  // 0..D_INNER-1
  const int b = blockIdx.y;
  __shared__ float BC[2 * D_STATE];

  float a[D_STATE], h[D_STATE];
  #pragma unroll
  for (int n = 0; n < D_STATE; ++n) {
    a[n] = -__expf(A_log[d * D_STATE + n]);
    h[n] = 0.f;
  }
  const float Dd = Dv[d];

  for (int t = 0; t < SEQ_LEN; ++t) {
    const size_t row = (size_t)b * SEQ_LEN + t;
    if (threadIdx.x < 2 * D_STATE)
      BC[threadIdx.x] = xdbl[row * XPROJ_N + DT_RANK + threadIdx.x];
    __syncthreads();
    const float dtv = dtb[row * D_INNER + d];
    const float xv  = xs[row * D_INNER + d];
    const float dx  = dtv * xv;
    float yv = 0.f;
    #pragma unroll
    for (int n = 0; n < D_STATE; ++n) {
      h[n] = __expf(dtv * a[n]) * h[n] + dx * BC[n];
      yv  += h[n] * BC[D_STATE + n];
    }
    yv += xv * Dd;
    const float zv = xz[row * (2 * D_INNER) + D_INNER + d];
    yv *= zv / (1.f + __expf(-zv));              // * SiLU(z)
    y[row * D_INNER + d] = (unsigned short)f2bf(yv);
    __syncthreads();
  }
}

// ---------------------------------------------------------------------------
extern "C" void kernel_launch(void* const* d_in, const int* in_sizes, int n_in,
                              void* d_out, int out_size, void* d_ws, size_t ws_size,
                              hipStream_t stream) {
  const float* x          = (const float*)d_in[0];
  const float* in_proj_w  = (const float*)d_in[1];
  const float* conv_w     = (const float*)d_in[2];
  const float* conv_b     = (const float*)d_in[3];
  const float* x_proj_w   = (const float*)d_in[4];
  const float* dt_proj_w  = (const float*)d_in[5];
  const float* dt_proj_b  = (const float*)d_in[6];
  const float* A_log      = (const float*)d_in[7];
  const float* Dv         = (const float*)d_in[8];
  const float* out_proj_w = (const float*)d_in[9];
  float* out = (float*)d_out;

  // ---- workspace layout ----
  // f32: xz | xs_f | x_dbl | dt          (~136 MB)
  float* xz   = (float*)d_ws;                               // MTOT*4096
  float* xs_f = xz   + (size_t)MTOT * 2 * D_INNER;          // MTOT*2048
  float* xdbl = xs_f + (size_t)MTOT * D_INNER;              // MTOT*96
  float* dtb  = xdbl + (size_t)MTOT * XPROJ_N;              // MTOT*2048
  // bf16 shadows (~39 MB)
  unsigned short* x_h    = (unsigned short*)(dtb + (size_t)MTOT * D_INNER);
  unsigned short* w1_h   = x_h  + (size_t)MTOT * D_MODEL;          // in_proj_w
  unsigned short* xs_h   = w1_h + (size_t)2 * D_INNER * D_MODEL;
  unsigned short* w2_h   = xs_h + (size_t)MTOT * D_INNER;          // x_proj_w
  unsigned short* xdbl_h = w2_h + (size_t)XPROJ_N * D_INNER;
  unsigned short* w3_h   = xdbl_h + (size_t)MTOT * XPROJ_N;        // dt_proj_w
  unsigned short* w4_h   = w3_h + (size_t)D_INNER * DT_RANK;       // out_proj_w
  unsigned short* y_h    = x_h;  // alias over x_h+w1_h (dead after in_proj)

  dim3 blk(256);
  auto cvt = [&](const float* s, unsigned short* d, size_t n) {
    int n4 = (int)(n / 4);
    f32_to_bf16_kernel<<<(n4 + 255) / 256, blk, 0, stream>>>(s, d, n4);
  };

  // 0) one-time f32 -> bf16 conversions (activations + weights)
  cvt(x,          x_h,  (size_t)MTOT * D_MODEL);
  cvt(in_proj_w,  w1_h, (size_t)2 * D_INNER * D_MODEL);
  cvt(x_proj_w,   w2_h, (size_t)XPROJ_N * D_INNER);
  cvt(dt_proj_w,  w3_h, (size_t)D_INNER * DT_RANK);
  cvt(out_proj_w, w4_h, (size_t)D_MODEL * D_INNER);

  // 1) in_proj: xz = x @ in_proj_w^T      (M=4096, N=4096, K=1024)
  gemm_bf16_wmma<0, 2 * D_INNER, D_MODEL, D_MODEL, D_MODEL, 2 * D_INNER>
      <<<dim3((2 * D_INNER) / BN, MTOT / BM), blk, 0, stream>>>(
      x_h, w1_h, xz, nullptr);

  // 2) depthwise conv + bias + SiLU -> xs (f32 + bf16)
  conv_silu_kernel<<<(MTOT * D_INNER) / 256, blk, 0, stream>>>(
      xz, conv_w, conv_b, xs_f, xs_h);

  // 3) x_proj: x_dbl = xs @ x_proj_w^T    (M=4096, N=96, K=2048)
  gemm_bf16_wmma<0, XPROJ_N, D_INNER, D_INNER, D_INNER, XPROJ_N>
      <<<dim3(1, MTOT / BM), blk, 0, stream>>>(
      xs_h, w2_h, xdbl, nullptr);

  // 3b) bf16 shadow of x_dbl (dt GEMM reads first 64 cols, lda=96)
  cvt(xdbl, xdbl_h, (size_t)MTOT * XPROJ_N);

  // 4) dt = softplus(x_dbl[:, :64] @ dt_proj_w^T + b)  (M=4096, N=2048, K=64)
  gemm_bf16_wmma<1, D_INNER, DT_RANK, XPROJ_N, DT_RANK, D_INNER>
      <<<dim3(D_INNER / BN, MTOT / BM), blk, 0, stream>>>(
      xdbl_h, w3_h, dtb, dt_proj_b);

  // 5) selective scan (+ xs*D, * SiLU(z)) -> y (bf16)
  scan_kernel<<<dim3(D_INNER / 256, BATCH), blk, 0, stream>>>(
      xdbl, dtb, xs_f, xz, A_log, Dv, y_h);

  // 6) out_proj: out = y @ out_proj_w^T   (M=4096, N=1024, K=2048)
  gemm_bf16_wmma<0, D_MODEL, D_INNER, D_INNER, D_INNER, D_MODEL>
      <<<dim3(D_MODEL / BN, MTOT / BM), blk, 0, stream>>>(
      y_h, w4_h, out, nullptr);
}
